// RNNModel_mit_topic_84258668413444
// MI455X (gfx1250) — matrix operand order repaired and checked by reading, not compiled, not run on hardware
//
#include <hip/hip_runtime.h>

// ---------------------------------------------------------------------------
// Topic-LSTM LM forward for MI455X (gfx1250, wave32, WMMA bf16).
//   T=128, B=32, D=H=1024, 4H=4096, L=3, TOPIC=128, V=16000
// Dense GEMMs -> V_WMMA_F32_16X16X32_BF16 (bf16 in, f32 acc), 128-row block
//               tiles for B-operand reuse.
// Recurrence  -> one persistent kernel per layer (grid barrier per step),
//               W_hh staged into LDS via GLOBAL_LOAD_ASYNC_TO_LDS_B128 and
//               kept resident for all 128 timesteps.
// ---------------------------------------------------------------------------

typedef __bf16 bf16_t;
typedef __attribute__((ext_vector_type(16))) __bf16 v16bf;
typedef __attribute__((ext_vector_type(8)))  __bf16 v8bf;
typedef __attribute__((ext_vector_type(8)))  float  v8f;

#define T_SEQ   128
#define BATCH   32
#define DIM     1024
#define HID     1024
#define FH      4096      // 4*HID
#define NLAYER  3
#define TOPICSZ 128
#define VOCAB   16000
#define RBLOCKS (HID / 16)   // 64 blocks in the recurrence kernel

__device__ __forceinline__ v16bf cat16(v8bf lo, v8bf hi) {
  return __builtin_shufflevector(lo, hi,
      0, 1, 2, 3, 4, 5, 6, 7, 8, 9, 10, 11, 12, 13, 14, 15);
}

__device__ __forceinline__ float sigf(float x) {
  return 1.0f / (1.0f + __expf(-x));
}

// ---------------------------------------------------------------------------
// f32 -> bf16 bulk conversion (grid-stride)
// ---------------------------------------------------------------------------
__global__ void cvt_f32_bf16_kernel(const float* __restrict__ src,
                                    bf16_t* __restrict__ dst, size_t n) {
  size_t i = (size_t)blockIdx.x * blockDim.x + threadIdx.x;
  size_t stride = (size_t)gridDim.x * blockDim.x;
  for (; i < n; i += stride) dst[i] = (bf16_t)src[i];
}

// ---------------------------------------------------------------------------
// Embedding gather, f32 table -> bf16 activations [T*B, D]
// ---------------------------------------------------------------------------
__global__ void embed_kernel(const int* __restrict__ tok,
                             const float* __restrict__ emb,
                             bf16_t* __restrict__ x) {
  size_t idx = (size_t)blockIdx.x * blockDim.x + threadIdx.x;
  if (idx >= (size_t)T_SEQ * BATCH * DIM) return;
  int row = (int)(idx / DIM);
  int col = (int)(idx % DIM);
  x[idx] = (bf16_t)emb[(size_t)tok[row] * DIM + col];
}

// ---------------------------------------------------------------------------
// Topic term: tg[b, n] = topic[b,:] . W_th[n,:] + b_g[n]   (B x 4H, K=128)
// ---------------------------------------------------------------------------
__global__ void topic_term_kernel(const float* __restrict__ topic,
                                  const float* __restrict__ Wth,
                                  const float* __restrict__ bg,
                                  float* __restrict__ tg) {
  int idx = blockIdx.x * blockDim.x + threadIdx.x;
  if (idx >= BATCH * FH) return;
  int b = idx / FH;
  int n = idx % FH;
  const float* tv = topic + (size_t)b * TOPICSZ;
  const float* w  = Wth + (size_t)n * TOPICSZ;
  float s = bg[n];
#pragma unroll 8
  for (int k = 0; k < TOPICSZ; ++k) s += tv[k] * w[k];
  tg[idx] = s;
}

// ---------------------------------------------------------------------------
// State init: h (bf16) and c (f32) from h0/c0 slices; zero the grid barrier
// ---------------------------------------------------------------------------
__global__ void init_state_kernel(const float* __restrict__ h0l,
                                  const float* __restrict__ c0l,
                                  bf16_t* __restrict__ hb,
                                  float* __restrict__ c,
                                  unsigned* __restrict__ bar) {
  int idx = blockIdx.x * blockDim.x + threadIdx.x;
  if (idx == 0) *bar = 0u;
  if (idx >= BATCH * HID) return;
  hb[idx] = (bf16_t)h0l[idx];
  c[idx]  = c0l[idx];
}

// ---------------------------------------------------------------------------
// Generic bf16 WMMA GEMM:  out[M,N] = A[M,K] * W[N,K]^T + extra
//   mode 0: extra[(m % 32) * N + n]   (topic broadcast over time)
//   mode 2: extra[n]                  (bias)
// Block = 128 threads = 4 waves covering a 128x64 output tile:
//   wave w owns M-tiles (blockIdx.y*128 + w*16) and (+64), and 4 N-tiles.
// Each B fragment is reused by 2 WMMAs; A fragment by 4. M must be a
// multiple of 128 (true for all callers: M = T*B = 4096).
// ---------------------------------------------------------------------------
__global__ __launch_bounds__(128)
void wmma_gemm_bf16(const bf16_t* __restrict__ A,
                    const bf16_t* __restrict__ W,
                    const float* __restrict__ extra,
                    float* __restrict__ out,
                    int M, int N, int K, int mode) {
  const int lane  = threadIdx.x & 31;
  const int wave  = threadIdx.x >> 5;
  const int mTile = blockIdx.y * 128 + wave * 16;   // second tile at +64
  const int nBase = blockIdx.x * 64;
  if (mTile + 64 > M) return;

  const int l16 = lane & 15;   // row (A) / col (B) within tile
  const int hl  = lane >> 4;   // half-wave selects K sub-range

  v8f acc[2][4] = {};

  const bf16_t* aRow0 = A + (size_t)(mTile + l16) * K;
  const bf16_t* aRow1 = aRow0 + (size_t)64 * K;

  for (int k0 = 0; k0 < K; k0 += 32) {
    // Two A 16x32 bf16 fragments (ISA 7.12.2 layout)
    v16bf af0 = cat16(*(const v8bf*)(aRow0 + k0 + hl * 8),
                      *(const v8bf*)(aRow0 + k0 + 16 + hl * 8));
    v16bf af1 = cat16(*(const v8bf*)(aRow1 + k0 + hl * 8),
                      *(const v8bf*)(aRow1 + k0 + 16 + hl * 8));

#pragma unroll
    for (int nt = 0; nt < 4; ++nt) {
      const int n = nBase + nt * 16 + l16;
      v16bf bfrag = cat16(*(const v8bf*)(W + (size_t)n * K + k0 + hl * 8),
                          *(const v8bf*)(W + (size_t)n * K + k0 + 16 + hl * 8));
      acc[0][nt] = __builtin_amdgcn_wmma_f32_16x16x32_bf16(
          false, af0, false, bfrag, (short)0, acc[0][nt], false, false);
      acc[1][nt] = __builtin_amdgcn_wmma_f32_16x16x32_bf16(
          false, af1, false, bfrag, (short)0, acc[1][nt], false, false);
    }
  }

#pragma unroll
  for (int mt = 0; mt < 2; ++mt) {
#pragma unroll
    for (int nt = 0; nt < 4; ++nt) {
      const int n = nBase + nt * 16 + l16;
#pragma unroll
      for (int r = 0; r < 8; ++r) {
        const int m = mTile + mt * 64 + r + hl * 8;  // C/D: VGPR r -> M=r+8*half
        float e = (mode == 0) ? extra[(size_t)(m & (BATCH - 1)) * N + n]
                              : extra[n];
        out[(size_t)m * N + n] = acc[mt][nt][r] + e;
      }
    }
  }
}

// ---------------------------------------------------------------------------
// Persistent LSTM recurrence, one launch per layer.
//   64 blocks, block bi owns h-columns [bi*16, bi*16+16).
//   Wave = gate (i,f,g,o): W_hh rows g*H + jBase + [0,16).
//   W_hh rows staged into dynamic LDS once (128 KB) via async-to-LDS DMA,
//   then reused from LDS for all 128 steps.
//   h ping-pongs between two global buffers; one grid barrier per step.
// ---------------------------------------------------------------------------
__global__ __launch_bounds__(128)
void lstm_recurrence_kernel(const bf16_t* __restrict__ Whh,  // [4H, H] bf16
                            const float* __restrict__ xg,    // [T, B, 4H]
                            bf16_t* __restrict__ hA,         // [B, H] bf16
                            bf16_t* __restrict__ hB,         // [B, H] bf16
                            float* __restrict__ c,           // [B, H] f32
                            bf16_t* __restrict__ xout,       // [T, B, H] bf16
                            unsigned* __restrict__ bar) {
  extern __shared__ bf16_t smemW[];            // [64][HID] = 128 KB
  __shared__ float gsh[4][BATCH][17];          // gate tiles (+1 pad)

  const int lane  = threadIdx.x & 31;
  const int wave  = threadIdx.x >> 5;          // gate index 0..3
  const int l16   = lane & 15;
  const int hl    = lane >> 4;
  const int jBase = blockIdx.x * 16;

  // ---- Stage this block's 64 W_hh rows into LDS via async DMA (b128) ----
  // LDS byte address = low 32 bits of the flat shared pointer (ISA 10.2).
  for (int idx = threadIdx.x; idx < (64 * HID) / 8; idx += 128) {
    int flat = idx * 8;
    int r = flat / HID;                        // 0..63 local row
    int k = flat % HID;
    int g = r >> 4;
    int j = r & 15;
    const bf16_t* src = Whh + ((size_t)(g * HID + jBase + j)) * HID + k;
    unsigned ldsAddr = (unsigned)(size_t)(smemW + (size_t)r * HID + k);
    unsigned long long gAddr = (unsigned long long)(size_t)src;
    asm volatile("global_load_async_to_lds_b128 %0, %1, off"
                 :: "v"(ldsAddr), "v"(gAddr) : "memory");
  }
  asm volatile("s_wait_asynccnt 0x0" ::: "memory");
  __syncthreads();

  const bf16_t* wRow = smemW + (size_t)(wave * 16 + l16) * HID;

  for (int t = 0; t < T_SEQ; ++t) {
    const bf16_t* hcur = (t & 1) ? hB : hA;
    bf16_t*       hnxt = (t & 1) ? hA : hB;

    // gates tile [32 x 16] for this wave's gate: acc0 rows 0-15, acc1 16-31
    v8f acc0 = v8f{};
    v8f acc1 = v8f{};
    for (int k0 = 0; k0 < HID; k0 += 32) {
      // B fragment from LDS (ds_load_b128 x2)
      v8bf blo = *(const v8bf*)(wRow + k0 + hl * 8);
      v8bf bhi = *(const v8bf*)(wRow + k0 + hl * 8 + 16);
      v16bf bfrag = cat16(blo, bhi);

      const bf16_t* a0 = hcur + (size_t)l16 * HID + k0 + hl * 8;
      v16bf af0 = cat16(*(const v8bf*)a0, *(const v8bf*)(a0 + 16));
      acc0 = __builtin_amdgcn_wmma_f32_16x16x32_bf16(
          false, af0, false, bfrag, (short)0, acc0, false, false);

      const bf16_t* a1 = hcur + (size_t)(16 + l16) * HID + k0 + hl * 8;
      v16bf af1 = cat16(*(const v8bf*)a1, *(const v8bf*)(a1 + 16));
      acc1 = __builtin_amdgcn_wmma_f32_16x16x32_bf16(
          false, af1, false, bfrag, (short)0, acc1, false, false);
    }

    // gates = acc + xg[t]; park in LDS for the pointwise combine
    const float* xgt = xg + (size_t)t * BATCH * FH + (size_t)wave * HID
                          + jBase + l16;
#pragma unroll
    for (int r = 0; r < 8; ++r) {
      int m0 = r + hl * 8;
      gsh[wave][m0][l16] = acc0[r] + xgt[(size_t)m0 * FH];
      int m1 = 16 + r + hl * 8;
      gsh[wave][m1][l16] = acc1[r] + xgt[(size_t)m1 * FH];
    }
    __syncthreads();

    // Pointwise LSTM update for the 32x16 (b, j) patch this block owns
    for (int e = threadIdx.x; e < BATCH * 16; e += 128) {
      int b  = e >> 4;
      int jj = e & 15;
      float ig = sigf(gsh[0][b][jj]);
      float fg = sigf(gsh[1][b][jj]);
      float gg = tanhf(gsh[2][b][jj]);
      float og = sigf(gsh[3][b][jj]);
      size_t cidx = (size_t)b * HID + jBase + jj;
      float cn = fg * c[cidx] + ig * gg;
      float hn = og * tanhf(cn);
      c[cidx] = cn;
      bf16_t h16 = (bf16_t)hn;
      hnxt[cidx] = h16;
      xout[(size_t)t * BATCH * HID + cidx] = h16;
    }
    __syncthreads();

    // Grid-wide barrier: make hnxt visible before any block starts step t+1
    if (threadIdx.x == 0) {
      __threadfence();
      atomicAdd(bar, 1u);
      unsigned target = (unsigned)gridDim.x * (unsigned)(t + 1);
      while (__hip_atomic_load(bar, __ATOMIC_ACQUIRE,
                               __HIP_MEMORY_SCOPE_AGENT) < target) {
        __builtin_amdgcn_s_sleep(2);
      }
    }
    __syncthreads();
  }
}

// ---------------------------------------------------------------------------
// Host-side orchestration (17 launches total)
// ---------------------------------------------------------------------------
extern "C" void kernel_launch(void* const* d_in, const int* in_sizes, int n_in,
                              void* d_out, int out_size, void* d_ws, size_t ws_size,
                              hipStream_t stream) {
  (void)in_sizes; (void)n_in; (void)out_size; (void)ws_size;

  const int*   tokens = (const int*)d_in[0];
  const float* topic  = (const float*)d_in[1];
  const float* h0     = (const float*)d_in[2];
  const float* c0     = (const float*)d_in[3];
  const float* emb    = (const float*)d_in[4];
  const float* W_ih   = (const float*)d_in[5];
  const float* W_hh   = (const float*)d_in[6];
  const float* W_th   = (const float*)d_in[7];
  const float* b_g    = (const float*)d_in[8];
  const float* dec_W  = (const float*)d_in[9];
  const float* dec_b  = (const float*)d_in[10];
  float* out = (float*)d_out;

  // Carve workspace (aligned to 256B)
  char* ws = (char*)d_ws;
  size_t off = 0;
  auto carve = [&](size_t bytes) -> void* {
    off = (off + 255) & ~(size_t)255;
    void* p = ws + off;
    off += bytes;
    return p;
  };

  bf16_t*   wih_b  = (bf16_t*)carve((size_t)NLAYER * FH * DIM * 2);
  bf16_t*   whh_b  = (bf16_t*)carve((size_t)NLAYER * FH * HID * 2);
  bf16_t*   decw_b = (bf16_t*)carve((size_t)VOCAB * HID * 2);
  bf16_t*   xA     = (bf16_t*)carve((size_t)T_SEQ * BATCH * DIM * 2);
  bf16_t*   xB     = (bf16_t*)carve((size_t)T_SEQ * BATCH * DIM * 2);
  float*    xg     = (float*)carve((size_t)T_SEQ * BATCH * FH * 4);
  float*    tg     = (float*)carve((size_t)BATCH * FH * 4);
  bf16_t*   hPing  = (bf16_t*)carve((size_t)BATCH * HID * 2);
  bf16_t*   hPong  = (bf16_t*)carve((size_t)BATCH * HID * 2);
  float*    cbuf   = (float*)carve((size_t)BATCH * HID * 4);
  unsigned* bar    = (unsigned*)carve(256);

  // 1) Weights -> bf16 (deterministic, every call)
  cvt_f32_bf16_kernel<<<2048, 256, 0, stream>>>(W_ih, wih_b,
      (size_t)NLAYER * FH * DIM);
  cvt_f32_bf16_kernel<<<2048, 256, 0, stream>>>(W_hh, whh_b,
      (size_t)NLAYER * FH * HID);
  cvt_f32_bf16_kernel<<<2048, 256, 0, stream>>>(dec_W, decw_b,
      (size_t)VOCAB * HID);

  // 2) Embedding gather -> bf16 activations
  {
    size_t n = (size_t)T_SEQ * BATCH * DIM;
    embed_kernel<<<(unsigned)((n + 255) / 256), 256, 0, stream>>>(tokens, emb, xA);
  }

  bf16_t* xin  = xA;
  bf16_t* xout = xB;

  for (int l = 0; l < NLAYER; ++l) {
    // Topic contribution + bias (constant over time)
    topic_term_kernel<<<(BATCH * FH + 255) / 256, 256, 0, stream>>>(
        topic, W_th + (size_t)l * FH * TOPICSZ, b_g + (size_t)l * FH, tg);

    // Batched gate GEMM over all timesteps: xg = xin @ W_ih^T + tg(broadcast)
    {
      dim3 grid(FH / 64, (T_SEQ * BATCH) / 128);
      wmma_gemm_bf16<<<grid, 128, 0, stream>>>(
          xin, wih_b + (size_t)l * FH * DIM, tg, xg,
          T_SEQ * BATCH, FH, DIM, /*mode=*/0);
    }

    // Init recurrent state + grid barrier counter
    init_state_kernel<<<(BATCH * HID + 255) / 256, 256, 0, stream>>>(
        h0 + (size_t)l * BATCH * HID, c0 + (size_t)l * BATCH * HID,
        hPing, cbuf, bar);

    // Entire 128-step recurrence in ONE persistent kernel
    lstm_recurrence_kernel<<<RBLOCKS, 128,
                             (size_t)64 * HID * sizeof(bf16_t), stream>>>(
        whh_b + (size_t)l * FH * HID, xg, hPing, hPong, cbuf, xout, bar);

    bf16_t* tmp = xin; xin = xout; xout = tmp;
  }

  // 3) Decoder: logits = x @ dec_W^T + dec_b  (4096 x 16000, K=1024)
  {
    dim3 grid(VOCAB / 64, (T_SEQ * BATCH) / 128);
    wmma_gemm_bf16<<<grid, 128, 0, stream>>>(
        xin, decw_b, dec_b, out, T_SEQ * BATCH, VOCAB, HID, /*mode=*/2);
  }
}
